// Attention_1108101563136
// MI455X (gfx1250) — compile-verified, matrix operand order checked by reading, not run on hardware
//
#include <hip/hip_runtime.h>

typedef __attribute__((ext_vector_type(16))) __bf16 v16bf;
typedef __attribute__((ext_vector_type(8)))  float  v8f;
typedef __attribute__((ext_vector_type(4)))  unsigned int u32x4;
typedef __attribute__((ext_vector_type(8)))  int          i32x8;
typedef __attribute__((ext_vector_type(4)))  int          i32x4;

#define DIMC   512
#define HEADS  8
#define DHEAD  64
#define SEQ    2048
#define BATCH  4
#define INNER  512

#define WMMA_BF16(A, B, C) \
  __builtin_amdgcn_wmma_f32_16x16x32_bf16(false, (A), false, (B), (short)0, (C), false, false)

// ---- fragment loaders -------------------------------------------------------
// A/B fragment (16-bit, K=32): lane holds row/col (lane&15); element e maps to
// k = (e<8 ? e : e+8) + 8*(lane>>4)  => two contiguous 8-element runs at
// offsets kb and kb+16, kb = 8*(lane>>4).

static __device__ __forceinline__ v16bf frag_from_f32(const float* __restrict__ row, int kb) {
  v16bf f;
#pragma unroll
  for (int i = 0; i < 8; ++i) f[i] = (__bf16)row[kb + i];
#pragma unroll
  for (int i = 0; i < 8; ++i) f[i + 8] = (__bf16)row[kb + 16 + i];
  return f;
}

static __device__ __forceinline__ v16bf frag_from_bf16(const __bf16* row, int kb) {
  v16bf f;
#pragma unroll
  for (int i = 0; i < 8; ++i) f[i] = row[kb + i];
#pragma unroll
  for (int i = 0; i < 8; ++i) f[i + 8] = row[kb + 16 + i];
  return f;
}

// ---- Tensor Data Mover: 2-D tile load into LDS ------------------------------
// Packs D# group0 (128b) / group1 (256b) per CDNA5 ISA
// (§8.3/§8.4): count=1, type=2("image"), data_size=1(2 bytes, bf16), 2-D tile,
// groups 2/3 zero. Optional LDS row padding (pad_amount DWORDs after every
// 2^(pi+1) DWORDs) to kill bank conflicts on the fragment reads.
// Units for dims/strides are elements (data_size units); addresses are bytes.
// Toolchain probe: 6-arg builtin (g0, g1, g2, g3, extra i32x8, cpol).
static __device__ __forceinline__ void tdm_load_2d_bf16(
    unsigned lds_byte_off, const void* gaddr,
    unsigned tile_d0, unsigned tile_d1,
    unsigned tensor_d0, unsigned tensor_d1,
    unsigned long long stride0,
    int pad_interval_code, int pad_amount_code) {
  unsigned long long ga = (unsigned long long)(uintptr_t)gaddr;
  u32x4 g0;
  g0[0] = 1u;                                             // count=1, user D#
  g0[1] = lds_byte_off;                                   // lds_addr
  g0[2] = (unsigned)(ga & 0xffffffffull);                 // global_addr[31:0]
  g0[3] = (unsigned)((ga >> 32) & 0x01ffffffull)          // global_addr[56:32]
        | (2u << 30);                                     // type=2
  i32x8 g1;
  g1[0] = (1 << 16)                                       // data_size=1 (2B)
        | (1 << 20)                                       // pad_enable
        | (pad_interval_code << 22)
        | (pad_amount_code << 25);
  g1[1] = (int)((tensor_d0 & 0xffffu) << 16);             // tensor_dim0[15:0]
  g1[2] = (int)((tensor_d0 >> 16) & 0xffffu)              // tensor_dim0[31:16]
        | (int)((tensor_d1 & 0xffffu) << 16);             // tensor_dim1[15:0]
  g1[3] = (int)((tensor_d1 >> 16) & 0xffffu)              // tensor_dim1[31:16]
        | (int)((tile_d0 & 0xffffu) << 16);               // tile_dim0
  g1[4] = (int)(tile_d1 & 0xffffu);                       // tile_dim1 (tile_dim2=0)
  g1[5] = (int)(stride0 & 0xffffffffull);                 // tensor_dim0_stride[31:0]
  g1[6] = (int)((stride0 >> 32) & 0xffffull);             // stride[47:32]; d1_stride=0
  g1[7] = 0;
  i32x4 z4 = {0, 0, 0, 0};
  i32x8 z8 = {0, 0, 0, 0, 0, 0, 0, 0};
  __builtin_amdgcn_tensor_load_to_lds(g0, g1, z4, z4, z8, 0);
}

// ---- kernel 1: QKV projection ----------------------------------------------
// qkv[m][f] = sum_c x[m][c] * w_qkv[f][c];  q scaled by dhead^-0.5.
// q,k scattered to [B,H,N,64] bf16; v stored TRANSPOSED to [B,H,64,N] bf16 so
// the attention kernel's P*V B-fragments read contiguous key-runs.
__global__ void __launch_bounds__(256)
qkv_proj_kernel(const float* __restrict__ x, const float* __restrict__ wqkv,
                __bf16* __restrict__ q, __bf16* __restrict__ k, __bf16* __restrict__ vt) {
  const int lane = threadIdx.x;
  const int wave = threadIdx.y;
  const int m0 = blockIdx.x * 256 + wave * 32;  // 32-row tile per wave
  const int f0 = blockIdx.y * 32;               // 32-col tile
  const int mlo = lane & 15;
  const int kb  = (lane >> 4) << 3;

  const float* arow0 = x    + (size_t)(m0 + mlo)      * DIMC;
  const float* arow1 = x    + (size_t)(m0 + 16 + mlo) * DIMC;
  const float* brow0 = wqkv + (size_t)(f0 + mlo)      * DIMC;
  const float* brow1 = wqkv + (size_t)(f0 + 16 + mlo) * DIMC;

  v8f acc00 = {}, acc01 = {}, acc10 = {}, acc11 = {};
#pragma unroll 4
  for (int k0 = 0; k0 < DIMC; k0 += 32) {
    __builtin_prefetch(arow0 + k0 + 64, 0, 0);
    __builtin_prefetch(brow0 + k0 + 64, 0, 0);
    v16bf a0 = frag_from_f32(arow0, k0 + kb);
    v16bf a1 = frag_from_f32(arow1, k0 + kb);
    v16bf b0 = frag_from_f32(brow0, k0 + kb);
    v16bf b1 = frag_from_f32(brow1, k0 + kb);
    acc00 = WMMA_BF16(a0, b0, acc00);
    acc01 = WMMA_BF16(a0, b1, acc01);
    acc10 = WMMA_BF16(a1, b0, acc10);
    acc11 = WMMA_BF16(a1, b1, acc11);
  }

  // C layout: m = r + 8*(lane>>4), n = lane&15
#pragma unroll
  for (int mi = 0; mi < 2; ++mi) {
#pragma unroll
    for (int ni = 0; ni < 2; ++ni) {
      v8f acc = (mi == 0) ? ((ni == 0) ? acc00 : acc01)
                          : ((ni == 0) ? acc10 : acc11);
#pragma unroll
      for (int r = 0; r < 8; ++r) {
        int row = m0 + mi * 16 + kb + r;     // 0..8191 (b*2048+n)
        int f   = f0 + ni * 16 + mlo;        // 0..1535
        int t   = f >> 9;                    // 0=q,1=k,2=v
        int rem = f & 511;
        int h   = rem >> 6;
        int d   = rem & 63;
        int b   = row >> 11;
        int n   = row & 2047;
        float val = acc[r];
        if (t == 0) {
          q[((size_t)((b << 3) + h) * SEQ + n) * DHEAD + d] = (__bf16)(val * 0.125f);
        } else if (t == 1) {
          k[((size_t)((b << 3) + h) * SEQ + n) * DHEAD + d] = (__bf16)val;
        } else {
          vt[((size_t)((b << 3) + h) * DHEAD + d) * SEQ + n] = (__bf16)val;
        }
      }
    }
  }
}

// ---- kernel 2: flash attention with TDM-staged K/V tiles --------------------
// 8 waves per WG share one (b,h); wave w owns query rows q0..q0+15. K tile
// (32x64) and V^T tile (64x32) are DMA'd into double-buffered LDS by the TDM
// (wave 0 issues, s_wait_tensorcnt, barrier). LDS rows padded via TDM
// pad_enable: K row stride 72 bf16 (pad 4 DW per 32 DW), V^T/P row stride 40
// bf16 (pad 4 DW per 16 DW) -> conflict-free half-wave fragment reads.
__global__ void __launch_bounds__(256)
flash_attn_kernel(const __bf16* __restrict__ q, const __bf16* __restrict__ k,
                  const __bf16* __restrict__ vt, __bf16* __restrict__ aout) {
  __shared__ __bf16 kbuf[2][32][72];   // keys x dhead (padded)
  __shared__ __bf16 vbuf[2][64][40];   // dhead x keys (padded)
  __shared__ __bf16 pbuf[8][16][40];   // per-wave P staging (padded)

  const int lane = threadIdx.x;
  const int wave = threadIdx.y;
  const int bh   = blockIdx.y;                   // b*8 + h
  const int q0   = blockIdx.x * 128 + wave * 16; // query row base
  const int mlo  = lane & 15;
  const int kb   = (lane >> 4) << 3;

  const __bf16* Q  = q  + (size_t)bh * SEQ * DHEAD;
  const __bf16* K  = k  + (size_t)bh * SEQ * DHEAD;
  const __bf16* Vt = vt + (size_t)bh * DHEAD * SEQ;

  // Q A-fragments for the full head dim (K=64 -> two K=32 steps)
  const __bf16* qrow = Q + (size_t)(q0 + mlo) * DHEAD;
  v16bf qa0 = frag_from_bf16(qrow, kb);
  v16bf qa1 = frag_from_bf16(qrow, 32 + kb);

  v8f o0 = {}, o1 = {}, o2 = {}, o3 = {};     // O accum: d-chunks 0..3 (16 each)
  float rowm[8], rowl[8];
#pragma unroll
  for (int r = 0; r < 8; ++r) { rowm[r] = -3.0e30f; rowl[r] = 0.0f; }

  // Prologue: TDM-load tile j0=0 into buffer 0.
  if (wave == 0) {
    tdm_load_2d_bf16((unsigned)(uintptr_t)&kbuf[0][0][0], K,
                     DHEAD, 32, DHEAD, SEQ, DHEAD, /*pi:32DW*/4, /*pa:4DW*/3);
    tdm_load_2d_bf16((unsigned)(uintptr_t)&vbuf[0][0][0], Vt,
                     32, DHEAD, SEQ, DHEAD, SEQ, /*pi:16DW*/3, /*pa:4DW*/3);
    __builtin_amdgcn_s_wait_tensorcnt(0);
  }
  __syncthreads();

  int cur = 0;
  for (int j0 = 0; j0 < SEQ; j0 += 32) {
    const int nxt = cur ^ 1;
    if (wave == 0 && j0 + 32 < SEQ) {   // async prefetch of next tile
      tdm_load_2d_bf16((unsigned)(uintptr_t)&kbuf[nxt][0][0],
                       K + (size_t)(j0 + 32) * DHEAD,
                       DHEAD, 32, DHEAD, SEQ, DHEAD, 4, 3);
      tdm_load_2d_bf16((unsigned)(uintptr_t)&vbuf[nxt][0][0],
                       Vt + (size_t)(j0 + 32),
                       32, DHEAD, SEQ, DHEAD, SEQ, 3, 3);
    }

    // S = Q * K^T : two 16x16 C tiles (keys j0..j0+15, j0+16..j0+31), K from LDS
    v8f s0 = {}, s1 = {};
    {
      const __bf16* kr0 = &kbuf[cur][mlo][0];
      const __bf16* kr1 = &kbuf[cur][16 + mlo][0];
      v16bf kb00 = frag_from_bf16(kr0, kb);
      v16bf kb01 = frag_from_bf16(kr0, 32 + kb);
      v16bf kb10 = frag_from_bf16(kr1, kb);
      v16bf kb11 = frag_from_bf16(kr1, 32 + kb);
      s0 = WMMA_BF16(qa0, kb00, s0);
      s0 = WMMA_BF16(qa1, kb01, s0);
      s1 = WMMA_BF16(qa0, kb10, s1);
      s1 = WMMA_BF16(qa1, kb11, s1);
    }

    // online softmax: row m = kb + r lives on 16 lanes (one half-wave)
#pragma unroll
    for (int r = 0; r < 8; ++r) {
      float tm = fmaxf(s0[r], s1[r]);
#pragma unroll
      for (int off = 1; off < 16; off <<= 1) tm = fmaxf(tm, __shfl_xor(tm, off, 16));
      float nm    = fmaxf(rowm[r], tm);
      float alpha = __expf(rowm[r] - nm);
      rowm[r] = nm;
      float p0 = __expf(s0[r] - nm);
      float p1 = __expf(s1[r] - nm);
      float ps = p0 + p1;
#pragma unroll
      for (int off = 1; off < 16; off <<= 1) ps += __shfl_xor(ps, off, 16);
      rowl[r] = rowl[r] * alpha + ps;
      int m = kb + r;
      pbuf[wave][m][mlo]      = (__bf16)p0;
      pbuf[wave][m][16 + mlo] = (__bf16)p1;
      o0[r] *= alpha; o1[r] *= alpha; o2[r] *= alpha; o3[r] *= alpha;
    }

    // P as A-fragment from LDS (same wave; DS ops are in-order)
    v16bf pa = frag_from_bf16(&pbuf[wave][mlo][0], kb);

    // O += P(16x32) * V(32x64): B-frag column n = d-dim; V^T rows are
    // contiguous key-runs in LDS.
#pragma unroll
    for (int c = 0; c < 4; ++c) {
      v16bf vb = frag_from_bf16(&vbuf[cur][c * 16 + mlo][0], kb);
      if (c == 0)      o0 = WMMA_BF16(pa, vb, o0);
      else if (c == 1) o1 = WMMA_BF16(pa, vb, o1);
      else if (c == 2) o2 = WMMA_BF16(pa, vb, o2);
      else             o3 = WMMA_BF16(pa, vb, o3);
    }

    if (wave == 0) __builtin_amdgcn_s_wait_tensorcnt(0);
    __syncthreads();
    cur = nxt;
  }

  // normalize + scatter to [B, N, H*64] bf16
  const int b = bh >> 3, h = bh & 7;
#pragma unroll
  for (int r = 0; r < 8; ++r) {
    float inv = 1.0f / rowl[r];
    int qrow_i = q0 + kb + r;
    size_t rowbase = ((size_t)b * SEQ + qrow_i) * (HEADS * DHEAD) + (size_t)h * DHEAD;
    aout[rowbase +  0 + mlo] = (__bf16)(o0[r] * inv);
    aout[rowbase + 16 + mlo] = (__bf16)(o1[r] * inv);
    aout[rowbase + 32 + mlo] = (__bf16)(o2[r] * inv);
    aout[rowbase + 48 + mlo] = (__bf16)(o3[r] * inv);
  }
}

// ---- kernel 3: output projection -------------------------------------------
// out[m][c] = sum_f aout[m][f] * w_out[c][f] + b_out[c]   (fp32 out)
__global__ void __launch_bounds__(256)
out_proj_kernel(const __bf16* __restrict__ a, const float* __restrict__ wout,
                const float* __restrict__ bias, float* __restrict__ out) {
  const int lane = threadIdx.x;
  const int wave = threadIdx.y;
  const int m0 = blockIdx.x * 256 + wave * 32;
  const int n0 = blockIdx.y * 32;
  const int mlo = lane & 15;
  const int kb  = (lane >> 4) << 3;

  const __bf16* ar0 = a    + (size_t)(m0 + mlo)      * INNER;
  const __bf16* ar1 = a    + (size_t)(m0 + 16 + mlo) * INNER;
  const float*  br0 = wout + (size_t)(n0 + mlo)      * INNER;
  const float*  br1 = wout + (size_t)(n0 + 16 + mlo) * INNER;

  v8f acc00 = {}, acc01 = {}, acc10 = {}, acc11 = {};
#pragma unroll 4
  for (int k0 = 0; k0 < INNER; k0 += 32) {
    __builtin_prefetch(ar0 + k0 + 64, 0, 0);
    v16bf a0 = frag_from_bf16(ar0, k0 + kb);
    v16bf a1 = frag_from_bf16(ar1, k0 + kb);
    v16bf b0 = frag_from_f32(br0, k0 + kb);
    v16bf b1 = frag_from_f32(br1, k0 + kb);
    acc00 = WMMA_BF16(a0, b0, acc00);
    acc01 = WMMA_BF16(a0, b1, acc01);
    acc10 = WMMA_BF16(a1, b0, acc10);
    acc11 = WMMA_BF16(a1, b1, acc11);
  }

#pragma unroll
  for (int r = 0; r < 8; ++r) {
    int row0 = m0 + kb + r;
    int row1 = m0 + 16 + kb + r;
    int col0 = n0 + mlo;
    int col1 = n0 + 16 + mlo;
    out[(size_t)row0 * DIMC + col0] = acc00[r] + bias[col0];
    out[(size_t)row0 * DIMC + col1] = acc01[r] + bias[col1];
    out[(size_t)row1 * DIMC + col0] = acc10[r] + bias[col0];
    out[(size_t)row1 * DIMC + col1] = acc11[r] + bias[col1];
  }
}

// ---- host launcher ----------------------------------------------------------
extern "C" void kernel_launch(void* const* d_in, const int* in_sizes, int n_in,
                              void* d_out, int out_size, void* d_ws, size_t ws_size,
                              hipStream_t stream) {
  (void)in_sizes; (void)n_in; (void)out_size; (void)ws_size;
  const float* x     = (const float*)d_in[0];
  const float* w_qkv = (const float*)d_in[1];
  const float* w_out = (const float*)d_in[2];
  const float* b_out = (const float*)d_in[3];
  float* out = (float*)d_out;

  const size_t qkv_elems = (size_t)BATCH * HEADS * SEQ * DHEAD;  // 4,194,304
  __bf16* q    = (__bf16*)d_ws;
  __bf16* k    = q + qkv_elems;
  __bf16* vt   = k + qkv_elems;    // V stored transposed [B,H,64,N]
  __bf16* aout = vt + qkv_elems;   // [B, N, 512] bf16; total ws use = 32 MB

  // QKV: M=8192 rows, N=1536 cols; wave = 32x32 tile, 8 waves/WG
  qkv_proj_kernel<<<dim3(32, 48), dim3(32, 8), 0, stream>>>(x, w_qkv, q, k, vt);
  // attention: grid = (qblocks of 128, B*H)
  flash_attn_kernel<<<dim3(SEQ / 128, BATCH * HEADS), dim3(32, 8), 0, stream>>>(q, k, vt, aout);
  // out proj: M=8192, N=512
  out_proj_kernel<<<dim3(32, 16), dim3(32, 8), 0, stream>>>(aout, w_out, b_out, out);
}
